// DynamicConv1D_24275155157555
// MI455X (gfx1250) — matrix operand order repaired and checked by reading, not compiled, not run on hardware
//
#include <hip/hip_runtime.h>

// ---- problem constants (from reference) ----
#define T_DIM   2048
#define B_DIM   16
#define C_DIM   512         // input channels (= GEMM K)
#define H_DIM   8
#define KW      7           // conv kernel width
#define NOUT    56          // H*KW (GEMM N)
#define NPAD    64          // N padded to 4 WMMA tiles
#define KSTRIDE 520         // LDS K-stride for W (bf16): 260 words, 260%64==4 -> conflict free
#define NROWS   (T_DIM * B_DIM)   // 32768 GEMM rows

typedef __attribute__((ext_vector_type(16))) __bf16 v16bf;
typedef __attribute__((ext_vector_type(8)))  __bf16 v8bf;
typedef __attribute__((ext_vector_type(8)))  float  v8f;

// =====================================================================
// Kernel A: logits = query @ W  (bf16 WMMA, f32 accum), then softmax(K=7)
//   grid = 256 blocks x 256 threads (8 waves); each wave = one 16-row tile
//   dynamic LDS: phase 1 = W^T bf16 [56][KSTRIDE]  (58240 B)
//                phase 2 = per-wave D scratch, 8 * 16*64 f32 (32 KB, aliased)
// =====================================================================
__global__ __launch_bounds__(256) void dynconv_gemm_softmax(
    const float* __restrict__ query,   // [NROWS][C_DIM]
    const float* __restrict__ W,       // [C_DIM][NOUT]
    float*       __restrict__ wgt)     // [NROWS][NOUT] softmaxed
{
    extern __shared__ char smem[];
    __bf16* Wt  = (__bf16*)smem;       // [56][KSTRIDE]
    float*  scr = (float*)smem;        // aliased after barrier

    const int tid  = threadIdx.x;
    const int lane = tid & 31;
    const int wave = tid >> 5;
    const int half = (lane >> 4) & 1;  // wave32 half
    const int mlan = lane & 15;

    // ---- stage W (f32 global) -> LDS bf16, transposed [n][k] ----
    for (int e = tid; e < NOUT * C_DIM; e += 256) {
        int k = e / NOUT;
        int n = e - k * NOUT;
        Wt[n * KSTRIDE + k] = (__bf16)W[k * NOUT + n];
    }
    __syncthreads();

    const int mtile = blockIdx.x * 8 + wave;           // 2048 tiles total
    const int row   = mtile * 16 + mlan;               // query row for this lane
    const float* qrow = query + (size_t)row * C_DIM;

    // B-fragment base pointers; columns >=56 clamp to 55 (results never read)
    const __bf16* wtn[4];
#pragma unroll
    for (int nt = 0; nt < 4; ++nt) {
        int n = nt * 16 + mlan;
        if (n > NOUT - 1) n = NOUT - 1;
        wtn[nt] = Wt + (size_t)n * KSTRIDE;
    }

    v8f acc[4];
#pragma unroll
    for (int nt = 0; nt < 4; ++nt)
#pragma unroll
        for (int i = 0; i < 8; ++i) acc[nt][i] = 0.0f;

    // ---- K loop: 16 steps of 32 ----
    for (int s = 0; s < 16; ++s) {
        const int kk0 = s * 32;
        // A fragment (ISA 16-bit A layout): lane half h holds
        //   K = {kk0+8h .. kk0+8h+7}  and  {kk0+16+8h .. kk0+16+8h+7}
        const float4 f0 = *(const float4*)(qrow + kk0 + half * 8);
        const float4 f1 = *(const float4*)(qrow + kk0 + half * 8 + 4);
        const float4 f2 = *(const float4*)(qrow + kk0 + 16 + half * 8);
        const float4 f3 = *(const float4*)(qrow + kk0 + 16 + half * 8 + 4);
        v16bf a;
        a[0]  = (__bf16)f0.x; a[1]  = (__bf16)f0.y; a[2]  = (__bf16)f0.z; a[3]  = (__bf16)f0.w;
        a[4]  = (__bf16)f1.x; a[5]  = (__bf16)f1.y; a[6]  = (__bf16)f1.z; a[7]  = (__bf16)f1.w;
        a[8]  = (__bf16)f2.x; a[9]  = (__bf16)f2.y; a[10] = (__bf16)f2.z; a[11] = (__bf16)f2.w;
        a[12] = (__bf16)f3.x; a[13] = (__bf16)f3.y; a[14] = (__bf16)f3.z; a[15] = (__bf16)f3.w;

#pragma unroll
        for (int nt = 0; nt < 4; ++nt) {
            // B fragment (matches A layout): lane half h holds
            //   K = {kk0+8h .. kk0+8h+7} and {kk0+16+8h .. kk0+16+8h+7} at column N
            const v8bf blo = *(const v8bf*)(wtn[nt] + kk0 + half * 8);
            const v8bf bhi = *(const v8bf*)(wtn[nt] + kk0 + 16 + half * 8);
            v16bf b;
#pragma unroll
            for (int i = 0; i < 8; ++i) { b[i] = blo[i]; b[8 + i] = bhi[i]; }
            acc[nt] = __builtin_amdgcn_wmma_f32_16x16x32_bf16(
                false, a, false, b, (short)0, acc[nt], false, false);
        }
    }

    __syncthreads();   // everyone done reading Wt -> reuse LDS as D scratch

    // D layout: lane holds N = nt*16 + mlan, VGPR v holds M = v + 8*half
    float* ws = scr + wave * (16 * NPAD);
#pragma unroll
    for (int nt = 0; nt < 4; ++nt)
#pragma unroll
        for (int v = 0; v < 8; ++v)
            ws[(v + 8 * half) * NPAD + nt * 16 + mlan] = acc[nt][v];
    __syncthreads();

    // ---- softmax over KW=7 per (row, head): 128 tasks per wave, 4/lane ----
#pragma unroll
    for (int it = 0; it < 4; ++it) {
        const int task = it * 32 + lane;
        const int m = task >> 3;       // row within tile
        const int h = task & 7;        // head
        float e[KW];
        float mx = -3.0e38f;
#pragma unroll
        for (int kk = 0; kk < KW; ++kk) {
            e[kk] = ws[m * NPAD + h * KW + kk];
            mx = fmaxf(mx, e[kk]);
        }
        float sum = 0.0f;
#pragma unroll
        for (int kk = 0; kk < KW; ++kk) { e[kk] = __expf(e[kk] - mx); sum += e[kk]; }
        const float inv = __frcp_rn(sum);
        const size_t obase = (size_t)(mtile * 16 + m) * NOUT + h * KW;
#pragma unroll
        for (int kk = 0; kk < KW; ++kk) wgt[obase + kk] = e[kk] * inv;
    }
}

// =====================================================================
// Kernel B: dynamic-conv combine.
//   out[t,b,c] = sum_kk xflat[7c+kk] * w[t,b, h(c)*7+kk]
//   where xflat = 7 SAME-padded time rows of x[:,b,:] laid out contiguously.
//   One block per (t,b); 512 threads; x staged via LDS (coalesced loads;
//   cross-block reuse of x rows hits the 192 MB L2).
// =====================================================================
__global__ __launch_bounds__(512) void dynconv_combine(
    const float* __restrict__ x,       // [T][B][C]
    const float* __restrict__ wgt,     // [NROWS][NOUT]
    float*       __restrict__ out)     // [T][B][C]
{
    __shared__ float xs[KW * C_DIM + NOUT];

    const int n   = blockIdx.x;        // (t,b) flat
    const int t   = n >> 4;
    const int b   = n & 15;
    const int tid = threadIdx.x;

#pragma unroll
    for (int k = 0; k < KW; ++k) {
        const int tt = t + k - 3;      // pad_before = (K-1)/2 = 3
        xs[k * C_DIM + tid] = (tt >= 0 && tt < T_DIM)
            ? x[((size_t)tt * B_DIM + b) * C_DIM + tid] : 0.0f;
    }
    if (tid < NOUT) xs[KW * C_DIM + tid] = wgt[(size_t)n * NOUT + tid];
    __syncthreads();

    const int c = tid;
    const int h = c >> 6;              // R = 64
    float acc = 0.0f;
#pragma unroll
    for (int kk = 0; kk < KW; ++kk)
        acc = fmaf(xs[7 * c + kk], xs[KW * C_DIM + h * KW + kk], acc);

    out[((size_t)t * B_DIM + b) * C_DIM + c] = acc;
}

// =====================================================================
extern "C" void kernel_launch(void* const* d_in, const int* in_sizes, int n_in,
                              void* d_out, int out_size, void* d_ws, size_t ws_size,
                              hipStream_t stream) {
    const float* x     = (const float*)d_in[0];   // (T,B,C) f32
    const float* query = (const float*)d_in[1];   // (T,B,C) f32
    const float* W     = (const float*)d_in[2];   // (C, H*K) f32
    float* out = (float*)d_out;
    float* wgt = (float*)d_ws;                    // 32768*56 f32 = 7.34 MB

    // Phase 1: GEMM(bf16 WMMA) + softmax -> weights
    const size_t ldsA = (size_t)NOUT * KSTRIDE * sizeof(__bf16); // 58240 B (>32KB scratch, aliased)
    hipLaunchKernelGGL(dynconv_gemm_softmax, dim3(NROWS / 16 / 8), dim3(256),
                       ldsA, stream, query, W, wgt);

    // Phase 2: dynamic conv combine
    hipLaunchKernelGGL(dynconv_combine, dim3(NROWS), dim3(512),
                       0, stream, x, wgt, out);
}
